// WOQ_Linear_Model_NoBias_3642132267641
// MI455X (gfx1250) — matrix operand order repaired and checked by reading, not compiled, not run on hardware
//
#include <hip/hip_runtime.h>
#include <stdint.h>

typedef __bf16  v8bf  __attribute__((ext_vector_type(8)));
typedef __bf16  v16bf __attribute__((ext_vector_type(16)));
typedef float   v8f   __attribute__((ext_vector_type(8)));
typedef int8_t  c16   __attribute__((ext_vector_type(16)));
typedef int     v4i   __attribute__((ext_vector_type(4)));

#define M_DIM 8192
#define K_DIM 4096
#define N_DIM 11008

#define BM 128
#define BN 128
#define BK 64
#define LDSW 72                 // bf16 elems per LDS row: 144B, 16B-aligned, skews banks
#define NTILES (K_DIM / BK)     // 64

#if defined(__has_builtin)
#  if __has_builtin(__builtin_amdgcn_global_load_async_to_lds_b128)
#    define USE_ASYNC_LDS 1
#  endif
#endif
#ifndef USE_ASYNC_LDS
#  define USE_ASYNC_LDS 0
#endif

#if USE_ASYNC_LDS
#  define WAIT_ASYNC() asm volatile("s_wait_asynccnt 0x0" ::: "memory")
#else
#  define WAIT_ASYNC() do {} while (0)
#endif

// Async builtin expects pointers-to-b128 (v4i) in explicit address spaces.
typedef __attribute__((address_space(1))) v4i* gptr_b128;
typedef __attribute__((address_space(3))) v4i* lptr_b128;

union V16 { v16bf v; v8bf h[2]; uint4 q[2]; };

__global__ __launch_bounds__(256)
void woq_bf16_wmma_kernel(const __bf16* __restrict__ X,
                          const int8_t* __restrict__ W,
                          const int8_t* __restrict__ ZP,
                          const __bf16* __restrict__ SC,
                          __bf16* __restrict__ OUTP)
{
    __shared__ __bf16 Xs[2][BM][LDSW];   // 2 x 18KB
    __shared__ __bf16 Ws[2][BN][LDSW];   // 2 x 18KB  (72KB total, OK vs 320KB/WGP)

    const int tid    = threadIdx.x;
    const int lane   = tid & 31;
    const int wave   = tid >> 5;
    const int laneLo = lane & 15;
    const int laneHi = lane >> 4;

    const int n0 = blockIdx.x * BN;   // 86 blocks
    const int m0 = blockIdx.y * BM;   // 64 blocks

    // 8 waves: 4 along M (32 rows each), 2 along N (64 cols each)
    const int waveM = (wave & 3) * 32;
    const int waveN = (wave >> 2) * 64;

    // ---- per-thread staging geometry (invariant over K) ----
    int xrow[4], xch[4];
#pragma unroll
    for (int i = 0; i < 4; ++i) { int l = tid + i * 256; xrow[i] = l >> 3; xch[i] = l & 7; }

    int wrow[2], wch[2];
    float sf[2], c1[2];
#pragma unroll
    for (int i = 0; i < 2; ++i) {
        int l = tid + i * 256; wrow[i] = l >> 2; wch[i] = l & 3;
        sf[i] = (float)SC[n0 + wrow[i]];              // per-output-channel scale
        float zpf = (float)ZP[n0 + wrow[i]];
        // perm trick: as_float(0x4B0000xx) = 2^23 + (w+128)  =>  (w - zp)*s = s*f + c1
        c1[i] = -sf[i] * (8388608.0f + 128.0f + zpf);
    }

    v8f acc[2][4];
    const v8f vz = {0.f,0.f,0.f,0.f,0.f,0.f,0.f,0.f};
#pragma unroll
    for (int a = 0; a < 2; ++a)
#pragma unroll
        for (int b = 0; b < 4; ++b) acc[a][b] = vz;

    c16 wq[2];   // in-flight W tile (int8) for the pipelined dequant

    // ---- X tile -> LDS (async DMA; no VGPR round-trip) ----
    auto stage_x = [&](int buf, int k) {
#pragma unroll
        for (int i = 0; i < 4; ++i) {
            const __bf16* src = X + (size_t)(m0 + xrow[i]) * K_DIM + k + xch[i] * 8;
#if USE_ASYNC_LDS
            __builtin_amdgcn_global_load_async_to_lds_b128(
                (gptr_b128)src, (lptr_b128)&Xs[buf][xrow[i]][xch[i] * 8], 0, 0);
#else
            uint4 v = *(const uint4*)src;
            *(uint4*)&Xs[buf][xrow[i]][xch[i] * 8] = v;
#endif
        }
    };

    // ---- W tile: global -> VGPR (issued early) ----
    auto wload = [&](int k) {
#pragma unroll
        for (int i = 0; i < 2; ++i)
            wq[i] = *(const c16*)(W + (size_t)(n0 + wrow[i]) * K_DIM + k + wch[i] * 16);
    };

    // ---- W tile: branchless perm-based dequant + store to LDS ----
    auto wstore = [&](int buf) {
#pragma unroll
        for (int i = 0; i < 2; ++i) {
            union { c16 c; uint32_t u[4]; } wu; wu.c = wq[i];
            V16 dq;
#pragma unroll
            for (int d = 0; d < 4; ++d) {
                uint32_t wx = wu.u[d] ^ 0x80808080u;   // bias int8 -> unsigned
#pragma unroll
                for (int j = 0; j < 4; ++j) {
                    // dst bytes [wx.byte(j), 0x00, 0x00, 0x4B]: sel 0x070404|j
                    uint32_t bits = __builtin_amdgcn_perm(0x4B000000u, wx, 0x07040400u | (uint32_t)j);
                    float f; __builtin_memcpy(&f, &bits, 4);
                    dq.v[d * 4 + j] = (__bf16)__builtin_fmaf(sf[i], f, c1[i]);
                }
            }
            *(uint4*)&Ws[buf][wrow[i]][wch[i] * 16]     = dq.q[0];
            *(uint4*)&Ws[buf][wrow[i]][wch[i] * 16 + 8] = dq.q[1];
        }
    };

    // ---- 16 WMMAs on one staged K-tile ----
    auto compute = [&](int buf) {
#pragma unroll
        for (int k0 = 0; k0 < BK; k0 += 32) {
            // A frag 16x32 bf16: lane<16 -> row=lane, K {0..7,16..23}; lane>=16 -> K {8..15,24..31}
            V16 af[2];
#pragma unroll
            for (int a = 0; a < 2; ++a) {
                int row = waveM + a * 16 + laneLo;
                int k   = k0 + laneHi * 8;
                af[a].h[0] = *(const v8bf*)&Xs[buf][row][k];
                af[a].h[1] = *(const v8bf*)&Xs[buf][row][k + 16];
            }
            // B frag 32x16 bf16: lane<16 -> col=lane, K {0..7,16..23}; lane>=16 -> K {8..15,24..31}
            V16 bfr[4];
#pragma unroll
            for (int b = 0; b < 4; ++b) {
                int row = waveN + b * 16 + laneLo;
                int k   = k0 + laneHi * 8;
                bfr[b].h[0] = *(const v8bf*)&Ws[buf][row][k];
                bfr[b].h[1] = *(const v8bf*)&Ws[buf][row][k + 16];
            }
#pragma unroll
            for (int a = 0; a < 2; ++a)
#pragma unroll
                for (int b = 0; b < 4; ++b)
                    acc[a][b] = __builtin_amdgcn_wmma_f32_16x16x32_bf16(
                        false, af[a].v, false, bfr[b].v,
                        (short)0, acc[a][b], false, false);
        }
    };

    // ================= software pipeline (double-buffered) =================
    stage_x(0, 0);
    wload(0);
    wstore(0);
    WAIT_ASYNC();
    __syncthreads();

    for (int kt = 0; kt < NTILES; ++kt) {
        const int cur = kt & 1;
        const bool have_next = (kt + 1 < NTILES);

        if (have_next) {
            stage_x(cur ^ 1, (kt + 1) * BK);   // async DMA into other buffer
            wload((kt + 1) * BK);              // W int8 in flight during compute
        }
        if (kt + 2 < NTILES)
            __builtin_prefetch(W + (size_t)(n0 + wrow[0]) * K_DIM + (kt + 2) * BK + wch[0] * 16, 0, 1);

        compute(cur);                          // 16 x v_wmma (hides W load latency)

        if (have_next) wstore(cur ^ 1);        // dequant + ds_store into other buffer

        WAIT_ASYNC();                          // async X writes visible
        __syncthreads();                       // ds stores visible (dscnt via barrier)
    }

    // ---- write back, non-temporal: keep 192MB L2 for reusable X/W stripes ----
#pragma unroll
    for (int a = 0; a < 2; ++a) {
#pragma unroll
        for (int b = 0; b < 4; ++b) {
            int col = n0 + waveN + b * 16 + laneLo;
#pragma unroll
            for (int i = 0; i < 8; ++i) {
                int row = m0 + waveM + a * 16 + laneHi * 8 + i;
                __builtin_nontemporal_store((__bf16)acc[a][b][i],
                                            OUTP + (size_t)row * N_DIM + col);
            }
        }
    }
}

extern "C" void kernel_launch(void* const* d_in, const int* in_sizes, int n_in,
                              void* d_out, int out_size, void* d_ws, size_t ws_size,
                              hipStream_t stream) {
    const __bf16*  x  = (const __bf16*)d_in[0];
    const int8_t*  w  = (const int8_t*)d_in[1];
    const int8_t*  zp = (const int8_t*)d_in[2];
    const __bf16*  sc = (const __bf16*)d_in[3];
    __bf16* out = (__bf16*)d_out;

    dim3 grid(N_DIM / BN, M_DIM / BM);   // 86 x 64 blocks
    woq_bf16_wmma_kernel<<<grid, 256, 0, stream>>>(x, w, zp, sc, out);
}
